// DRMM_56899726737593
// MI455X (gfx1250) — matrix operand order repaired and checked by reading, not compiled, not run on hardware
//
#include <hip/hip_runtime.h>
#include <hip/hip_bf16.h>
#include <stdint.h>

typedef __attribute__((ext_vector_type(16))) _Float16 v16h;
typedef __attribute__((ext_vector_type(8)))  _Float16 v8h;
typedef __attribute__((ext_vector_type(8)))  float    v8f;

#define NBINS 30
#define EPS   1e-5f
#define B_    64
#define Q_    32
#define D_    4096
#define E_    300
#define KP    320      // E padded to multiple of 32 (10 K-chunks of 32)
#define DT    32       // doc rows per workgroup

// ---------------- zero the global histogram ----------------
__global__ void drmm_zero(float* __restrict__ p, int n) {
    int i = blockIdx.x * 256 + threadIdx.x;
    if (i < n) p[i] = 0.0f;
}

// ------------- normalize query rows -> f16 [B*Q][KP] -------------
__global__ __launch_bounds__(64) void drmm_prep_q(const float* __restrict__ query,
                                                  _Float16* __restrict__ qh) {
    const int row = blockIdx.x;            // b*Q_ + q
    const int t   = threadIdx.x;           // 64 threads
    const float* qr = query + (size_t)row * E_;
    float s = 0.0f;
    for (int k = t; k < E_; k += 64) { float v = qr[k]; s += v * v; }
    __shared__ float red[64];
    red[t] = s;
    __syncthreads();
    for (int off = 32; off > 0; off >>= 1) {
        if (t < off) red[t] += red[t + off];
        __syncthreads();
    }
    const float rn = rsqrtf(red[0]);
    _Float16* qo = qh + (size_t)row * KP;
    for (int k = t; k < KP; k += 64)
        qo[k] = (k < E_) ? (_Float16)(qr[k] * rn) : (_Float16)0.0f;
}

// ------------- main: normalize docs, WMMA interaction, histogram -------------
__global__ __launch_bounds__(128) void drmm_main(const float* __restrict__ doc,
                                                 const _Float16* __restrict__ qh,
                                                 float* __restrict__ hist) {
    __shared__ _Float16 qsh[Q_][KP];        // 20480 B
    __shared__ _Float16 dnh[DT][KP];        // 20480 B
    __shared__ float    rnorm[DT];
    __shared__ unsigned lhist[Q_][NBINS];   // 3840 B

    const int tid  = threadIdx.x;
    const int wave = tid >> 5;
    const int lane = tid & 31;
    const int b    = blockIdx.x >> 7;       // / (D_/DT)
    const int dt   = blockIdx.x & 127;
    const int d0   = dt * DT;
    const float*    db = doc + ((size_t)b * D_ + d0) * E_;
    const _Float16* qb = qh + (size_t)b * Q_ * KP;

    // load query tile (32*320 halves = 5120 dwords) into LDS
    {
        const uint32_t* src = (const uint32_t*)qb;
        uint32_t*       dst = (uint32_t*)&qsh[0][0];
        for (int i = tid; i < Q_ * KP / 2; i += 128) dst[i] = src[i];
    }
    for (int i = tid; i < Q_ * NBINS; i += 128) ((unsigned*)lhist)[i] = 0u;

    // pass 1: per-row L2 norms (wave w owns rows w*8 .. w*8+7)
    for (int r = wave * 8; r < wave * 8 + 8; ++r) {
        const float* rp = db + (size_t)r * E_;
        float s = 0.0f;
        for (int k = lane; k < E_; k += 32) { float v = rp[k]; s += v * v; }
        #pragma unroll
        for (int off = 16; off > 0; off >>= 1) s += __shfl_xor(s, off, 32);
        if (lane == 0) rnorm[r] = rsqrtf(s);
    }
    __syncthreads();

    // pass 2: normalized f16 doc rows into LDS (second global read hits L2)
    for (int r = wave * 8; r < wave * 8 + 8; ++r) {
        const float* rp = db + (size_t)r * E_;
        const float  rn = rnorm[r];
        for (int k = lane; k < KP; k += 32)
            dnh[r][k] = (k < E_) ? (_Float16)(rp[k] * rn) : (_Float16)0.0f;
    }
    __syncthreads();

    // WMMA: wave -> (m_tile, n_tile); 4 waves cover 2 M-tiles x 2 N-tiles
    const int m_tile = wave & 1;
    const int n_tile = wave >> 1;
    const int half   = lane >> 4;
    const int l15    = lane & 15;
    const int m = m_tile * 16 + l15;        // A: lane holds row M
    const int n = n_tile * 16 + l15;        // B: lane holds col N

    v8f acc = {};
    #pragma unroll
    for (int c = 0; c < KP / 32; ++c) {
        const int kb = c * 32;
        // A fragment (16x32): lanes 0-15 K=kb..kb+15, lanes 16-31 K=kb+16..kb+31
        v16h a = *(const v16h*)&qsh[m][kb + 16 * half];
        // B fragment (32x16): lanes 0-15 K=kb..kb+15, lanes 16-31 K=kb+16..kb+31
        v16h bf = *(const v16h*)&dnh[n][kb + 16 * half];
        acc = __builtin_amdgcn_wmma_f32_16x16x32_f16(
            false, a, false, bf, (short)0, acc, false, false);
    }

    // bin accumulator: C VGPR r -> M = r + 8*half  (N = lane%16 = this doc row)
    #pragma unroll
    for (int r = 0; r < 8; ++r) {
        const int q   = m_tile * 16 + half * 8 + r;
        const float x = acc[r];
        int bin = (int)floorf((x + 1.0f) * (NBINS * 0.5f));
        bin = bin < 0 ? 0 : (bin > NBINS - 1 ? NBINS - 1 : bin);
        atomicAdd(&lhist[q][bin], 1u);
    }
    __syncthreads();

    // flush LDS histogram to global (float counts, matching reference dtype)
    float* hb = hist + (size_t)b * Q_ * NBINS;
    for (int i = tid; i < Q_ * NBINS; i += 128) {
        const unsigned v = ((unsigned*)lhist)[i];
        if (v) atomicAdd(&hb[i], (float)v);
    }
}

// ------------- finalize: log1p + FFN + gate + masked softmax + reduce -------------
__global__ __launch_bounds__(32) void drmm_finalize(
    const float* __restrict__ hist, const float* __restrict__ q_idf,
    const float* __restrict__ qmask,
    const float* __restrict__ w1, const float* __restrict__ b1,
    const float* __restrict__ w2, const float* __restrict__ b2,
    const float* __restrict__ w3, const float* __restrict__ b3,
    const float* __restrict__ gw, const float* __restrict__ gb,
    float* __restrict__ out) {
    const int b = blockIdx.x;
    const int q = threadIdx.x;              // 32 lanes = one wave32
    const float* h = hist + ((size_t)b * Q_ + q) * NBINS;

    float hl[NBINS];
    #pragma unroll
    for (int k = 0; k < NBINS; ++k) hl[k] = log1pf(h[k]);

    float z1[5];
    #pragma unroll
    for (int j = 0; j < 5; ++j) {
        float s = b1[j];
        #pragma unroll
        for (int k = 0; k < NBINS; ++k) s += hl[k] * w1[j * NBINS + k];
        z1[j] = tanhf(s);
    }
    float s2 = b2[0];
    #pragma unroll
    for (int j = 0; j < 5; ++j) s2 += z1[j] * w2[j];
    const float z2 = tanhf(s2);
    const float z3 = tanhf(z2 * w3[0] + b3[0]);

    const float g  = tanhf(q_idf[b * Q_ + q] * gw[0] + gb[0]);
    float ex  = __expf(g) * qmask[b * Q_ + q];
    float num = z3 * ex;
    #pragma unroll
    for (int off = 16; off > 0; off >>= 1) {
        ex  += __shfl_xor(ex,  off, 32);
        num += __shfl_xor(num, off, 32);
    }
    if (q == 0) out[b] = num / (ex + EPS);
}

extern "C" void kernel_launch(void* const* d_in, const int* in_sizes, int n_in,
                              void* d_out, int out_size, void* d_ws, size_t ws_size,
                              hipStream_t stream) {
    const float* query = (const float*)d_in[0];
    const float* qmask = (const float*)d_in[1];
    const float* doc   = (const float*)d_in[2];
    // d_in[3] = document_mask: unused by the reference
    const float* q_idf = (const float*)d_in[4];
    const float* w1 = (const float*)d_in[5];
    const float* b1 = (const float*)d_in[6];
    const float* w2 = (const float*)d_in[7];
    const float* b2 = (const float*)d_in[8];
    const float* w3 = (const float*)d_in[9];
    const float* b3 = (const float*)d_in[10];
    const float* gw = (const float*)d_in[11];
    const float* gb = (const float*)d_in[12];
    float* out = (float*)d_out;

    // workspace layout: [qh: B*Q*KP f16][hist: B*Q*NBINS f32]
    _Float16* qh   = (_Float16*)d_ws;
    float*    hist = (float*)((char*)d_ws + (size_t)B_ * Q_ * KP * sizeof(_Float16));
    const int nhist = B_ * Q_ * NBINS;      // 61440

    drmm_zero<<<(nhist + 255) / 256, 256, 0, stream>>>(hist, nhist);
    drmm_prep_q<<<B_ * Q_, 64, 0, stream>>>(query, qh);
    drmm_main<<<B_ * (D_ / DT), 128, 0, stream>>>(doc, qh, hist);
    drmm_finalize<<<B_, 32, 0, stream>>>(hist, q_idf, qmask,
                                         w1, b1, w2, b2, w3, b3, gw, gb, out);
}
